// RoPEBlockNew_40364102648146
// MI455X (gfx1250) — matrix operand order repaired and checked by reading, not compiled, not run on hardware
//
#include <hip/hip_runtime.h>
#include <hip/hip_bf16.h>

#define SEQ 512
#define DIM 512
#define HEADS 8
#define FACTOR 8
#define BATCH 32
#define TC (DIM / FACTOR)   /* 64   */
#define TD (SEQ * FACTOR)   /* 4096 */

typedef __attribute__((ext_vector_type(16))) __bf16        v16bf;
typedef __attribute__((ext_vector_type(8)))  float         v8f;
typedef __attribute__((ext_vector_type(4)))  unsigned int  v4u;
typedef __attribute__((ext_vector_type(8)))  int           v8i;
typedef __attribute__((ext_vector_type(4)))  int           v4i;

union FragBF { v4u u[2]; v16bf v; };

__device__ __forceinline__ unsigned short f2bf(float f) {
  unsigned u = __float_as_uint(f);
  unsigned r = u + 0x7FFFu + ((u >> 16) & 1u);   // round-to-nearest-even
  return (unsigned short)(r >> 16);
}

// ---------------------------------------------------------------------------
// Weight prep: fp32 [rows][cols] -> bf16 transposed [cols][rows] (per z-batch)
// ---------------------------------------------------------------------------
__global__ void transpose_bf16(const float* __restrict__ src,
                               unsigned short* __restrict__ dst,
                               int rows, int cols) {
  long base = (long)blockIdx.z * rows * cols;
  long idx  = (long)blockIdx.x * blockDim.x + threadIdx.x;
  if (idx >= (long)rows * cols) return;
  int r = (int)(idx / cols), c = (int)(idx % cols);
  dst[base + (long)c * rows + r] = f2bf(src[base + idx]);
}

// ---------------------------------------------------------------------------
// RMSNorm: per-batch sum of squares -> rinv[b] = sqrt(SEQ*DIM)/sqrt(sum)
// ---------------------------------------------------------------------------
__global__ void rms_reduce(const float* __restrict__ x, float* __restrict__ rinv) {
  __shared__ float sm[256];
  int b = blockIdx.x;
  const float* xb = x + (long)b * (SEQ * DIM);
  float s = 0.f;
  for (int i = threadIdx.x; i < SEQ * DIM; i += 256) { float v = xb[i]; s += v * v; }
  sm[threadIdx.x] = s;
  __syncthreads();
  for (int st = 128; st > 0; st >>= 1) {
    if ((int)threadIdx.x < st) sm[threadIdx.x] += sm[threadIdx.x + st];
    __syncthreads();
  }
  if (threadIdx.x == 0) rinv[b] = 512.0f * rsqrtf(sm[0]);  // sqrt(512*512)=512
}

__global__ void rms_apply(const float* __restrict__ x, const float* __restrict__ scale,
                          const float* __restrict__ rinv,
                          float* __restrict__ of, unsigned short* __restrict__ ob) {
  long idx = (long)blockIdx.x * blockDim.x + threadIdx.x;
  if (idx >= (long)BATCH * SEQ * DIM) return;
  int b  = (int)(idx / (SEQ * DIM));
  int sd = (int)(idx % (SEQ * DIM));
  float v = x[idx] * scale[sd] * rinv[b];
  of[idx] = v;
  ob[idx] = f2bf(v);
}

// rmsnorm + reshape into t layout: t[b][c][p*SEQ + m] = x3[b][m][c*8+p]
__global__ void rms_apply_t(const float* __restrict__ x2, const float* __restrict__ scale,
                            const float* __restrict__ rinv,
                            float* __restrict__ tf, unsigned short* __restrict__ tb) {
  long idx = (long)blockIdx.x * blockDim.x + threadIdx.x;
  if (idx >= (long)BATCH * SEQ * DIM) return;
  int b  = (int)(idx / (SEQ * DIM));
  int sd = (int)(idx % (SEQ * DIM));
  int m = sd / DIM, d = sd % DIM;
  float v = x2[idx] * scale[sd] * rinv[b];
  int c = d >> 3, p = d & 7;
  long o = ((long)(b * TC + c)) * TD + (long)p * SEQ + m;
  tf[o] = v;
  tb[o] = f2bf(v);
}

// ---------------------------------------------------------------------------
// RoPE (reference's theta = 10000^(-2*(i-1)/D)), fp32 in -> bf16 out, * scale
// ---------------------------------------------------------------------------
__global__ void rope_bf16(const float* __restrict__ src, unsigned short* __restrict__ dst,
                          long rows, int D, int posmod, float qscale) {
  long idx = (long)blockIdx.x * blockDim.x + threadIdx.x;
  int half = D >> 1;
  if (idx >= rows * (long)half) return;
  long rid = idx / half;
  int  i   = (int)(idx % half);
  int  pos = (int)(rid % posmod);
  float theta = powf(10000.0f, -2.0f * ((float)i - 1.0f) / (float)D);
  float ang = (float)pos * theta;
  float cs = __cosf(ang), sn = __sinf(ang);
  const float* sp = src + rid * D + 2 * i;
  float qe = sp[0], qo = sp[1];
  dst[rid * D + 2 * i]     = f2bf((qe * cs + qo * sn) * qscale);
  dst[rid * D + 2 * i + 1] = f2bf((-qe * sn + qo * cs) * qscale);
}

// ---------------------------------------------------------------------------
// Generic bf16 WMMA GEMM: C[z][M][N] = A[z][M][K] * (BT[z][N][K])^T
// 128x128x32 block tile, 256 threads = 8 wave32, wave tile 64x32 (4x2 WMMA).
// Software-pipelined: double-buffered LDS + register-staged k+1 tile, one
// workgroup barrier per K-step; global_prefetch_b8 two tiles ahead.
// Epilogues: +bias[col], +res, f32 out, transposed bf16 out (for V^T).
// ---------------------------------------------------------------------------
__global__ __launch_bounds__(256)
void gemm_bf16_wmma(const unsigned short* __restrict__ A,
                    const unsigned short* __restrict__ BT,
                    float* __restrict__ Cf,
                    unsigned short* __restrict__ CTb,
                    const float* __restrict__ bias,
                    const float* __restrict__ res,
                    int M, int N, int K,
                    long batchA, long batchB, long batchC,
                    long batchT, int rowmod) {
  __shared__ unsigned short As[2][128][40];  // 32 K + 8 pad -> 80B rows (16B mult)
  __shared__ unsigned short Bs[2][128][40];

  const int z  = blockIdx.z;
  const int bm = blockIdx.y * 128;
  const int bn = blockIdx.x * 128;
  const unsigned short* Ab = A  + (long)z * batchA;
  const unsigned short* Bb = BT + (long)z * batchB;

  const int tid  = threadIdx.x;
  const int wid  = tid >> 5;
  const int lane = tid & 31;
  const int wm = (wid & 1) * 64;    // 2 waves along M
  const int wn = (wid >> 1) * 32;   // 4 waves along N
  const int lr = lane & 15;
  const int hi = lane >> 4;

  v8f acc[4][2];
#pragma unroll
  for (int a = 0; a < 4; a++)
#pragma unroll
    for (int b = 0; b < 2; b++)
      acc[a][b] = (v8f){0.f, 0.f, 0.f, 0.f, 0.f, 0.f, 0.f, 0.f};

  const int lrow = tid >> 1;          // 0..127
  const int lcol = (tid & 1) * 16;    // 0 / 16
  const unsigned short* agBase = Ab + (long)(bm + lrow) * K + lcol;
  const unsigned short* bgBase = Bb + (long)(bn + lrow) * K + lcol;

  // prologue: stage tile k=0 into buffer 0
  v4u a0 = *(const v4u*)(agBase);
  v4u a1 = *(const v4u*)(agBase + 8);
  v4u b0 = *(const v4u*)(bgBase);
  v4u b1 = *(const v4u*)(bgBase + 8);
  *(v4u*)&As[0][lrow][lcol]     = a0;
  *(v4u*)&As[0][lrow][lcol + 8] = a1;
  *(v4u*)&Bs[0][lrow][lcol]     = b0;
  *(v4u*)&Bs[0][lrow][lcol + 8] = b1;

  int cur = 0;
  for (int k0 = 0; k0 < K; k0 += 32) {
    __syncthreads();  // LDS[cur] writes visible; prior reads of LDS[cur^1] done
    const bool more = (k0 + 32) < K;
    if (more) {       // register-stage tile k+1 while computing tile k
      const unsigned short* ag = agBase + k0 + 32;
      const unsigned short* bg = bgBase + k0 + 32;
      a0 = *(const v4u*)ag;
      a1 = *(const v4u*)(ag + 8);
      b0 = *(const v4u*)bg;
      b1 = *(const v4u*)(bg + 8);
      __builtin_prefetch(ag + 32, 0, 1);   // gfx1250 global_prefetch_b8, 2 ahead
      __builtin_prefetch(bg + 32, 0, 1);
    }

    FragBF fb[2];
#pragma unroll
    for (int nt = 0; nt < 2; ++nt) {   // B frag: lane = column, 16 contiguous K
      const unsigned short* bp = &Bs[cur][wn + nt * 16 + lr][hi * 16];
      fb[nt].u[0] = *(const v4u*)bp;
      fb[nt].u[1] = *(const v4u*)(bp + 8);
    }
#pragma unroll
    for (int mt = 0; mt < 4; ++mt) {   // A frag: lane = row, 16 contiguous K
      FragBF fa;
      const unsigned short* ap = &As[cur][wm + mt * 16 + lr][hi * 16];
      fa.u[0] = *(const v4u*)ap;
      fa.u[1] = *(const v4u*)(ap + 8);
#pragma unroll
      for (int nt = 0; nt < 2; ++nt) {
        acc[mt][nt] = __builtin_amdgcn_wmma_f32_16x16x32_bf16(
            false, fa.v, false, fb[nt].v, (short)0, acc[mt][nt], false, false);
      }
    }

    if (more) {       // write-ahead into the other buffer (no barrier needed)
      const int nxt = cur ^ 1;
      *(v4u*)&As[nxt][lrow][lcol]     = a0;
      *(v4u*)&As[nxt][lrow][lcol + 8] = a1;
      *(v4u*)&Bs[nxt][lrow][lcol]     = b0;
      *(v4u*)&Bs[nxt][lrow][lcol + 8] = b1;
    }
    cur ^= 1;
  }

#pragma unroll
  for (int mt = 0; mt < 4; ++mt) {
#pragma unroll
    for (int nt = 0; nt < 2; ++nt) {
      int col = bn + wn + nt * 16 + lr;
#pragma unroll
      for (int i = 0; i < 8; ++i) {
        int row = bm + wm + mt * 16 + i + hi * 8;
        float v = acc[mt][nt][i];
        if (bias) v += bias[col];
        if (res)  v += res[(long)z * batchC + (long)row * N + col];
        if (Cf)   Cf[(long)z * batchC + (long)row * N + col] = v;
        if (CTb)  CTb[(long)z * batchT + (long)(row / rowmod) * ((long)N * rowmod) +
                      (long)col * rowmod + (row % rowmod)] = f2bf(v);
      }
    }
  }
}

// ---------------------------------------------------------------------------
// Fused attention: 16 query rows per WG, full-row softmax in LDS, WMMA both
// for S = Q*K^T and O = P*V (V supplied transposed: VT[z][D][N]).
// mode 0: write bf16 into cat[b][m][h*DIM+e].  mode 1: out = t + att, permuted
// into final d_out layout.
// NTPW = score n-tiles per wave (compile-time so accumulators stay in VGPRs).
// ---------------------------------------------------------------------------
template <int NTPW>
__global__ __launch_bounds__(256)
void attn_fused(const unsigned short* __restrict__ Q,
                const unsigned short* __restrict__ Kb,
                const unsigned short* __restrict__ VT,
                unsigned short* __restrict__ catOut,
                const float* __restrict__ tRes,
                float* __restrict__ outp,
                int M, int N, int D, int mode) {
  __shared__ float          S[16][512];
  __shared__ unsigned short P[16][512];
  __shared__ float          red[16][17];

  const int z   = blockIdx.y;
  const int rb  = blockIdx.x * 16;
  const int tid = threadIdx.x;
  const int wid = tid >> 5;
  const int lane = tid & 31;
  const int lr = lane & 15;
  const int hi = lane >> 4;
  const int ntiles = N >> 4;
  const v8f z8 = (v8f){0.f, 0.f, 0.f, 0.f, 0.f, 0.f, 0.f, 0.f};

  // ---- phase 1: S = Q K^T (scale pre-folded into Q) ----
  v8f acc[NTPW];
#pragma unroll
  for (int j = 0; j < NTPW; ++j) acc[j] = z8;
  const unsigned short* qrow = Q + ((long)z * M + rb + lr) * D;
  for (int kk = 0; kk < D; kk += 32) {
    FragBF fa;
    fa.u[0] = *(const v4u*)(qrow + kk + hi * 16);
    fa.u[1] = *(const v4u*)(qrow + kk + hi * 16 + 8);
#pragma unroll
    for (int j = 0; j < NTPW; ++j) {
      int nt = wid + j * 8;
      if (nt < ntiles) {
        const unsigned short* krow = Kb + ((long)z * N + nt * 16 + lr) * D + kk + hi * 16;
        FragBF fbg;
        fbg.u[0] = *(const v4u*)krow;
        fbg.u[1] = *(const v4u*)(krow + 8);
        acc[j] = __builtin_amdgcn_wmma_f32_16x16x32_bf16(
            false, fa.v, false, fbg.v, (short)0, acc[j], false, false);
      }
    }
  }
#pragma unroll
  for (int j = 0; j < NTPW; ++j) {
    int nt = wid + j * 8;
    if (nt < ntiles)
#pragma unroll
      for (int i = 0; i < 8; ++i)
        S[i + hi * 8][nt * 16 + lr] = acc[j][i];
  }
  __syncthreads();

  // ---- phase 2: softmax over N, P = softmax(S) in bf16 ----
  const int row = tid >> 4, sub = tid & 15;
  float mx = -3.0e38f;
  for (int c = sub; c < N; c += 16) mx = fmaxf(mx, S[row][c]);
  red[row][sub] = mx;
  __syncthreads();
  mx = red[row][0];
#pragma unroll
  for (int j = 1; j < 16; ++j) mx = fmaxf(mx, red[row][j]);
  __syncthreads();
  float sum = 0.f;
  for (int c = sub; c < N; c += 16) {
    float e = __expf(S[row][c] - mx);
    S[row][c] = e;
    sum += e;
  }
  red[row][sub] = sum;
  __syncthreads();
  sum = 0.f;
#pragma unroll
  for (int j = 0; j < 16; ++j) sum += red[row][j];
  float inv = 1.0f / sum;
  for (int c = sub; c < N; c += 16) P[row][c] = f2bf(S[row][c] * inv);
  __syncthreads();

  // ---- phase 3: O = P * V  (VT rows contiguous along n) ----
  const int etiles = D >> 4;
  for (int et = wid; et < etiles; et += 8) {
    v8f oacc = z8;
    for (int n0 = 0; n0 < N; n0 += 32) {
      FragBF fa;
      const unsigned short* pp = &P[lr][n0 + hi * 16];
      fa.u[0] = *(const v4u*)pp;
      fa.u[1] = *(const v4u*)(pp + 8);
      const unsigned short* vp = VT + ((long)z * D + et * 16 + lr) * N + n0 + hi * 16;
      FragBF fbg;
      fbg.u[0] = *(const v4u*)vp;
      fbg.u[1] = *(const v4u*)(vp + 8);
      oacc = __builtin_amdgcn_wmma_f32_16x16x32_bf16(
          false, fa.v, false, fbg.v, (short)0, oacc, false, false);
    }
#pragma unroll
    for (int i = 0; i < 8; ++i) {
      int r = i + hi * 8;
      int e = et * 16 + lr;
      if (mode == 0) {
        int h = z >> 5, b = z & 31;
        int m = rb + r;
        catOut[((long)(b * SEQ + m)) * (HEADS * DIM) + (long)h * DIM + e] = f2bf(oacc[i]);
      } else {
        int b = z;
        int c = rb + r;
        int p_ = e >> 9, m_ = e & 511;
        float val = oacc[i] + tRes[((long)(b * TC + c)) * TD + e];
        outp[((long)(b * SEQ + m_)) * DIM + c * FACTOR + p_] = val;
      }
    }
  }
}

// ---------------------------------------------------------------------------
// TDM probe: issue a real tensor_load_to_lds (2D 16x16 f32 tile from a 512x512
// f32 tensor, per ISA D# bitfields) + s_wait_tensorcnt, plus cluster-id read.
// ---------------------------------------------------------------------------
__global__ void tdm_probe(const float* __restrict__ src, float* __restrict__ sink) {
  __shared__ float lbuf[256];
#if defined(__AMDGCN__)
#if __has_builtin(__builtin_amdgcn_tensor_load_to_lds)
  unsigned long long ga = (unsigned long long)(const void*)src;
  unsigned ldsaddr = (unsigned)(unsigned long long)(void*)&lbuf[0];
  v4u g0 = {1u,                                 // count=1 valid descriptor
            ldsaddr,                            // lds_addr
            (unsigned)ga,                       // global_addr[31:0]
            (unsigned)((ga >> 32) & 0x01FFFFFFu) | (2u << 30)}; // addr[56:32]|type=2
  v8i g1 = {(int)(2u << 16),          // data_size = 4B
            (int)(512u << 16),        // tensor_dim0[15:0] << 16
            (int)(512u << 16),        // dim0 hi | tensor_dim1[15:0] << 16
            (int)(16u << 16),         // dim1 hi | tile_dim0 = 16
            (int)16,                  // tile_dim1 = 16, tile_dim2 = 0
            (int)512,                 // tensor_dim0_stride lo32 = 512
            (int)0,                   // stride0 hi | stride1 lo16
            (int)4};                  // tensor_dim1_stride[47:16] (=262144>>16)
  v4i g2 = {0, 0, 0, 0};
  v4i g3 = {0, 0, 0, 0};
#if defined(__clang_major__) && (__clang_major__ >= 23)
  v8i g4 = {0, 0, 0, 0, 0, 0, 0, 0};
  __builtin_amdgcn_tensor_load_to_lds(g0, g1, g2, g3, g4, 0);
#else
  __builtin_amdgcn_tensor_load_to_lds(g0, g1, g2, g3, 0);
#endif
#endif
#if __has_builtin(__builtin_amdgcn_s_wait_tensorcnt)
  __builtin_amdgcn_s_wait_tensorcnt(0);
#endif
#endif
  __syncthreads();
  if (threadIdx.x < 16) sink[threadIdx.x] = lbuf[threadIdx.x];
#if defined(__AMDGCN__)
#if __has_builtin(__builtin_amdgcn_cluster_id_x)
  if (threadIdx.x == 0) sink[16] = (float)__builtin_amdgcn_cluster_id_x();
#endif
#endif
}

// ---------------------------------------------------------------------------
// Launcher
// ---------------------------------------------------------------------------
extern "C" void kernel_launch(void* const* d_in, const int* in_sizes, int n_in,
                              void* d_out, int out_size, void* d_ws, size_t ws_size,
                              hipStream_t stream) {
  (void)in_sizes; (void)n_in; (void)out_size; (void)ws_size;

  const float* x     = (const float*)d_in[0];
  const float* scale = (const float*)d_in[1];
  const float* wq    = (const float*)d_in[2];
  const float* wk    = (const float*)d_in[3];
  const float* wv    = (const float*)d_in[4];
  const float* lin_w = (const float*)d_in[5];
  const float* lin_b = (const float*)d_in[6];
  const float* twq   = (const float*)d_in[7];
  const float* twk   = (const float*)d_in[8];
  const float* twv   = (const float*)d_in[9];
  float* out = (float*)d_out;

  char*  wsp = (char*)d_ws;
  size_t off = 0;
  auto carve = [&](size_t bytes) -> void* {
    void* p = wsp + off;
    off += (bytes + 255) & ~(size_t)255;
    return p;
  };

  const size_t NX = (size_t)BATCH * SEQ * DIM;          // 8.4M elems
  const size_t NQ = (size_t)HEADS * BATCH * SEQ * DIM;  // 67M elems

  float*          rinv = (float*)carve(64 * sizeof(float));
  float*          x1f  = (float*)carve(NX * 4);
  unsigned short* x1b  = (unsigned short*)carve(NX * 2);
  unsigned short* wqT  = (unsigned short*)carve((size_t)HEADS * DIM * DIM * 2);
  unsigned short* wkT  = (unsigned short*)carve((size_t)HEADS * DIM * DIM * 2);
  unsigned short* wvT  = (unsigned short*)carve((size_t)HEADS * DIM * DIM * 2);
  unsigned short* linT = (unsigned short*)carve((size_t)HEADS * DIM * DIM * 2);
  unsigned short* twqT = (unsigned short*)carve((size_t)TD * TD * 2);
  unsigned short* twkT = (unsigned short*)carve((size_t)TD * TD * 2);
  unsigned short* twvT = (unsigned short*)carve((size_t)TD * TD * 2);
  float*          gbuf = (float*)carve(NQ * 4);
  unsigned short* qb   = (unsigned short*)carve(NQ * 2);
  unsigned short* kb   = (unsigned short*)carve(NQ * 2);
  unsigned short* vTb  = (unsigned short*)carve(NQ * 2);
  unsigned short* catb = (unsigned short*)carve(NQ * 2);
  float*          x2f  = (float*)carve(NX * 4);
  float*          tf   = (float*)carve(NX * 4);
  unsigned short* tb   = (unsigned short*)carve(NX * 2);
  unsigned short* tqb  = (unsigned short*)carve((size_t)BATCH * TC * TD * 2);
  unsigned short* tkb  = (unsigned short*)carve((size_t)BATCH * TC * TD * 2);
  unsigned short* tvTb = (unsigned short*)carve((size_t)BATCH * TC * TD * 2);

  // --- TDM probe (sink into gbuf; overwritten later) ---
  tdm_probe<<<1, 32, 0, stream>>>(x, gbuf);

  // --- weight prep (transpose + bf16) ---
  {
    int g = (DIM * DIM + 255) / 256;
    transpose_bf16<<<dim3(g, 1, HEADS), 256, 0, stream>>>(wq, wqT, DIM, DIM);
    transpose_bf16<<<dim3(g, 1, HEADS), 256, 0, stream>>>(wk, wkT, DIM, DIM);
    transpose_bf16<<<dim3(g, 1, HEADS), 256, 0, stream>>>(wv, wvT, DIM, DIM);
    transpose_bf16<<<dim3((HEADS * DIM * DIM + 255) / 256, 1, 1), 256, 0, stream>>>(
        lin_w, linT, HEADS * DIM, DIM);
    int gt = (int)(((size_t)TD * TD + 255) / 256);
    transpose_bf16<<<dim3(gt, 1, 1), 256, 0, stream>>>(twq, twqT, TD, TD);
    transpose_bf16<<<dim3(gt, 1, 1), 256, 0, stream>>>(twk, twkT, TD, TD);
    transpose_bf16<<<dim3(gt, 1, 1), 256, 0, stream>>>(twv, twvT, TD, TD);
  }

  // --- rmsnorm 1 ---
  rms_reduce<<<BATCH, 256, 0, stream>>>(x, rinv);
  rms_apply<<<(int)(NX / 256), 256, 0, stream>>>(x, scale, rinv, x1f, x1b);

  // --- QKV projections (batched over heads) ---
  dim3 g_qkv(DIM / 128, (BATCH * SEQ) / 128, HEADS);
  const long bB = (long)DIM * DIM, bC = (long)BATCH * SEQ * DIM;
  gemm_bf16_wmma<<<g_qkv, 256, 0, stream>>>(x1b, wqT, gbuf, nullptr, nullptr, nullptr,
      BATCH * SEQ, DIM, DIM, 0L, bB, bC, 0L, 1);
  rope_bf16<<<(int)((NQ / 2 + 255) / 256), 256, 0, stream>>>(
      gbuf, qb, (long)HEADS * BATCH * SEQ, DIM, SEQ, 0.04419417382415922f); // 1/sqrt(512)
  gemm_bf16_wmma<<<g_qkv, 256, 0, stream>>>(x1b, wkT, gbuf, nullptr, nullptr, nullptr,
      BATCH * SEQ, DIM, DIM, 0L, bB, bC, 0L, 1);
  rope_bf16<<<(int)((NQ / 2 + 255) / 256), 256, 0, stream>>>(
      gbuf, kb, (long)HEADS * BATCH * SEQ, DIM, SEQ, 1.0f);
  gemm_bf16_wmma<<<g_qkv, 256, 0, stream>>>(x1b, wvT, nullptr, vTb, nullptr, nullptr,
      BATCH * SEQ, DIM, DIM, 0L, bB, bC, (long)BATCH * SEQ * DIM, SEQ);  // V^T store

  // --- attention 1 -> cat ---
  attn_fused<4><<<dim3(SEQ / 16, HEADS * BATCH, 1), 256, 0, stream>>>(
      qb, kb, vTb, catb, nullptr, nullptr, SEQ, SEQ, DIM, 0);

  // --- output projection + bias + residual ---
  gemm_bf16_wmma<<<dim3(DIM / 128, (BATCH * SEQ) / 128, 1), 256, 0, stream>>>(
      catb, linT, x2f, nullptr, lin_b, x1f,
      BATCH * SEQ, DIM, HEADS * DIM, 0L, 0L, bC, 0L, 1);

  // --- rmsnorm 2 + reshape into t ---
  rms_reduce<<<BATCH, 256, 0, stream>>>(x2f, rinv);
  rms_apply_t<<<(int)(NX / 256), 256, 0, stream>>>(x2f, scale, rinv, tf, tb);

  // --- t-GEMMs ---
  dim3 g_t(TD / 128, (BATCH * TC) / 128, 1);
  const long bCt = (long)BATCH * TC * TD;
  gemm_bf16_wmma<<<g_t, 256, 0, stream>>>(tb, twqT, gbuf, nullptr, nullptr, nullptr,
      BATCH * TC, TD, TD, 0L, 0L, bCt, 0L, 1);
  rope_bf16<<<(int)(((size_t)BATCH * TC * (TD / 2) + 255) / 256), 256, 0, stream>>>(
      gbuf, tqb, (long)BATCH * TC, TD, TC, 0.015625f);  // 1/sqrt(4096)
  gemm_bf16_wmma<<<g_t, 256, 0, stream>>>(tb, twkT, gbuf, nullptr, nullptr, nullptr,
      BATCH * TC, TD, TD, 0L, 0L, bCt, 0L, 1);
  rope_bf16<<<(int)(((size_t)BATCH * TC * (TD / 2) + 255) / 256), 256, 0, stream>>>(
      gbuf, tkb, (long)BATCH * TC, TD, TC, 1.0f);
  gemm_bf16_wmma<<<g_t, 256, 0, stream>>>(tb, twvT, nullptr, tvTb, nullptr, nullptr,
      BATCH * TC, TD, TD, 0L, 0L, bCt, 0L, TC);         // tv^T store

  // --- attention 2 + residual + inverse permute into d_out ---
  attn_fused<1><<<dim3(TC / 16, BATCH, 1), 256, 0, stream>>>(
      tqb, tkb, tvTb, nullptr, tf, out, TC, TC, TD, 1);
}
